// HardNegativeCrossAttention_26637387169889
// MI455X (gfx1250) — matrix operand order repaired and checked by reading, not compiled, not run on hardware
//
#include <hip/hip_runtime.h>
#include <math.h>

typedef __attribute__((ext_vector_type(16))) __bf16 v16bf;
typedef __attribute__((ext_vector_type(8)))  __bf16 v8bf;
typedef __attribute__((ext_vector_type(8)))  float  v8f;

#define WMMA_BF16(a, b, c) \
  __builtin_amdgcn_wmma_f32_16x16x32_bf16(false, (a), false, (b), (short)0, (c), false, false)

// ---------------------------------------------------------------------------
// Fragment loaders (wave32 WMMA 16x16x32 bf16 layouts, CDNA5 ISA 7.12.2)
// A (16x32): lane m=lane&15 holds K runs [hi*8 .. hi*8+7] and [16+hi*8 ..]
// B (32x16): lane col=lane&15 holds K runs [hi*8 .. hi*8+7] and [16+hi*8 ..]
// ---------------------------------------------------------------------------

// A-fragment from bf16 row-major [rows x ld] (k-contiguous rows)
static __device__ __forceinline__ v16bf load_a_bf16(const __bf16* p, int ld, int row0, int k0) {
  const int lane = threadIdx.x & 31;
  const __bf16* r = p + (size_t)(row0 + (lane & 15)) * ld + k0 + (lane >> 4) * 8;
  union { v16bf v; v8bf h[2]; } u;
  u.h[0] = *(const v8bf*)(r);       // global_load_b128 / ds_load_b128
  u.h[1] = *(const v8bf*)(r + 16);
  return u.v;
}

// A-fragment from fp32 row-major [rows x ld], converted to bf16 in-register
static __device__ __forceinline__ v16bf load_a_f32(const float* p, int ld, int row0, int k0) {
  const int lane = threadIdx.x & 31;
  const float* r = p + (size_t)(row0 + (lane & 15)) * ld + k0 + (lane >> 4) * 8;
  v16bf v;
#pragma unroll
  for (int i = 0; i < 8; ++i) v[i] = (__bf16)r[i];
#pragma unroll
  for (int i = 0; i < 8; ++i) v[8 + i] = (__bf16)r[16 + i];
  return v;
}

// B-fragment (32x16) from a K-contiguous source: element [col][k], stride ld
static __device__ __forceinline__ v16bf load_b_kmaj(const __bf16* p, int ld, int col0, int k0) {
  const int lane = threadIdx.x & 31;
  const __bf16* r = p + (size_t)(col0 + (lane & 15)) * ld + k0 + (lane >> 4) * 8;
  union { v16bf v; v8bf h[2]; } u;
  u.h[0] = *(const v8bf*)(r);
  u.h[1] = *(const v8bf*)(r + 16);
  return u.v;
}

// ---------------------------------------------------------------------------
// Kernel 1: 1x1-conv projection  Out = W(MxK=128) * X(128xN) + bias
// TRANS_OUT=1 -> Out stored (N,M) bf16 (k-contiguous for later A/B frags)
// TRANS_OUT=0 -> Out stored (M,N) bf16 (token-contiguous, used for V)
// Block: 256 thr (8 waves), tile 128(M) x 64(tokens), X tile staged to LDS^T.
// All dims compile-time -> address math folds into instruction immediates.
// ---------------------------------------------------------------------------
template <bool TRANS_OUT, int M, int N>
__global__ __launch_bounds__(256) void proj_kernel(
    const float* __restrict__ W, const float* __restrict__ bias,
    const float* __restrict__ Xall, __bf16* __restrict__ Out) {
  constexpr int K = 128;
  const int b = blockIdx.z;
  const float* X = Xall + (size_t)b * K * N;
  __bf16* Ob = Out + (size_t)b * M * N;
  const int n0 = blockIdx.x * 64;

  __shared__ __bf16 sXT[64 * K];  // [token][c], bf16, 16 KB
  for (int idx = threadIdx.x; idx < K * 16; idx += 256) {
    const int c = idx >> 4;
    const int n = (idx & 15) * 4;
    const float4 f = *(const float4*)(X + (size_t)c * N + n0 + n);
    sXT[(n + 0) * K + c] = (__bf16)f.x;
    sXT[(n + 1) * K + c] = (__bf16)f.y;
    sXT[(n + 2) * K + c] = (__bf16)f.z;
    sXT[(n + 3) * K + c] = (__bf16)f.w;
  }
  __syncthreads();

  const int m0   = blockIdx.y * 128 + (threadIdx.x >> 5) * 16;
  const int lane = threadIdx.x & 31, lo = lane & 15, hi = lane >> 4;

  v8f acc[4] = {};
#pragma unroll
  for (int kk = 0; kk < K; kk += 32) {
    const v16bf a = load_a_f32(W, K, m0, kk);  // weight frag, reused 4x
#pragma unroll
    for (int t = 0; t < 4; ++t) {
      const v16bf bf = load_b_kmaj(sXT, K, t * 16, kk);
      acc[t] = WMMA_BF16(a, bf, acc[t]);
    }
  }

  float bv[8];
#pragma unroll
  for (int r = 0; r < 8; ++r) bv[r] = bias[m0 + 8 * hi + r];

#pragma unroll
  for (int t = 0; t < 4; ++t) {
    const int n = n0 + t * 16 + lo;
    if (TRANS_OUT) {
      v8bf pk;
#pragma unroll
      for (int r = 0; r < 8; ++r) pk[r] = (__bf16)(acc[t][r] + bv[r]);
      *(v8bf*)(Ob + (size_t)n * M + m0 + 8 * hi) = pk;  // one b128 store
    } else {
#pragma unroll
      for (int r = 0; r < 8; ++r)
        Ob[(size_t)(m0 + 8 * hi + r) * N + n] = (__bf16)(acc[t][r] + bv[r]);
    }
  }
}

// ---------------------------------------------------------------------------
// Kernel 2: flash attention.  Q,Kt: (N,128) bf16; V: (768,N) bf16.
// Block = 32 queries, 64-key chunks.
//   Q A-fragments are loop-invariant -> hoisted into registers (4 x v16bf).
//   Phase A: all 8 waves compute one 16x16 S tile each (2 qt x 4 kt).
//   Phase B: softmax parallelized 8 threads/row with shfl_xor reductions.
//   Phase C: 2-deep manual pipeline of V fragments (>=4 loads in flight);
//            V tile offsets are compile-time immediates (N constexpr).
// ---------------------------------------------------------------------------
template <int N>
__global__ __launch_bounds__(256) void attn_kernel(
    const __bf16* __restrict__ Qall, const __bf16* __restrict__ Ktall,
    const __bf16* __restrict__ Vall, __bf16* __restrict__ Oall) {
  constexpr int C = 128, E = 768;
  const int b = blockIdx.y;
  const __bf16* Q  = Qall  + (size_t)b * N * C;
  const __bf16* Kt = Ktall + (size_t)b * N * C;
  const __bf16* V  = Vall  + (size_t)b * E * N;
  __bf16*       O  = Oall  + (size_t)b * N * E;

  const int q0 = blockIdx.x * 32;
  const int tid = threadIdx.x;
  const int w = tid >> 5, lane = tid & 31, lo = lane & 15, hi = lane >> 4;
  const int qt = w & 1;               // PV phase: which 16-row q tile
  const int e_base = (w >> 1) * 192;  // PV phase: this wave's E slice
  const int sqt = w & 1, skt = w >> 1;  // S phase: 2 x 4 tile assignment

  __shared__ float  sS[32 * 64];      // raw logits, 8 KB
  __shared__ __bf16 sP[32 * 64];      // exp weights, 4 KB
  __shared__ float  sM[32], sL[32], sAlpha[32], sLinv[32];

  v8f acc[12] = {};                   // 16 x 192 fp32 accumulator

  // Hoisted, loop-invariant Q fragments for this wave's S-tile row
  v16bf qa[4];
#pragma unroll
  for (int kq = 0; kq < 4; ++kq)
    qa[kq] = load_a_bf16(Q, C, q0 + sqt * 16, kq * 32);

  if (tid < 32) { sM[tid] = -1e30f; sL[tid] = 0.0f; }
  __syncthreads();

  for (int kc = 0; kc < N; kc += 64) {
    if (kc + 64 < N) {                // warm next chunk toward L2/WGP$
      __builtin_prefetch(Kt + (size_t)(kc + 64 + skt * 16) * C, 0, 1);
      __builtin_prefetch(V + kc + 64, 0, 1);
    }
    // --- Phase A: S = Q K^T, one 16x16 tile per wave (all 8 waves) ---
    {
      v8f s = {};
#pragma unroll
      for (int kq = 0; kq < 4; ++kq) {
        const v16bf bb = load_b_kmaj(Kt, C, kc + skt * 16, kq * 32);
        s = WMMA_BF16(qa[kq], bb, s);
      }
#pragma unroll
      for (int r = 0; r < 8; ++r)
        sS[(sqt * 16 + 8 * hi + r) * 64 + skt * 16 + lo] = s[r];
    }
    __syncthreads();
    // --- Phase B: online softmax, 8 threads per query row ---
    {
      const float scale = 0.08838834764831845f;  // 1/sqrt(128)
      const int row = tid >> 3;
      const int j8  = (tid & 7) * 8;
      float sv[8];
#pragma unroll
      for (int j = 0; j < 8; ++j) sv[j] = sS[row * 64 + j8 + j] * scale;
      float mx = sv[0];
#pragma unroll
      for (int j = 1; j < 8; ++j) mx = fmaxf(mx, sv[j]);
#pragma unroll
      for (int d = 1; d < 8; d <<= 1) mx = fmaxf(mx, __shfl_xor(mx, d, 32));
      const float m_old = sM[row];
      const float m_new = fmaxf(m_old, mx);
      v8bf pk;
      float sum = 0.0f;
#pragma unroll
      for (int j = 0; j < 8; ++j) {
        const float p = __expf(sv[j] - m_new);
        sum += p;
        pk[j] = (__bf16)p;
      }
      *(v8bf*)(sP + row * 64 + j8) = pk;   // one ds_store_b128
#pragma unroll
      for (int d = 1; d < 8; d <<= 1) sum += __shfl_xor(sum, d, 32);
      if ((tid & 7) == 0) {
        const float alpha = __expf(m_old - m_new);
        sAlpha[row] = alpha;
        sM[row] = m_new;
        sL[row] = sL[row] * alpha + sum;
      }
    }
    __syncthreads();
    // --- Phase C: rescale accumulator, O += P * V (all 8 waves) ---
    {
      float al[8];
#pragma unroll
      for (int r = 0; r < 8; ++r) al[r] = sAlpha[qt * 16 + 8 * hi + r];
#pragma unroll
      for (int t = 0; t < 12; ++t) {
#pragma unroll
        for (int r = 0; r < 8; ++r) acc[t][r] *= al[r];
      }
      const v16bf pa0 = load_a_bf16(sP, 64, qt * 16, 0);   // keys kc..kc+31
      const v16bf pa1 = load_a_bf16(sP, 64, qt * 16, 32);  // keys kc+32..kc+63
      // 2-deep software pipeline: tile t+1 loads issued before tile t WMMAs
      v16bf vb0 = load_b_kmaj(V, N, e_base, kc);
      v16bf vb1 = load_b_kmaj(V, N, e_base, kc + 32);
#pragma unroll
      for (int t = 0; t < 12; ++t) {
        v16bf nb0 = vb0, nb1 = vb1;
        if (t < 11) {
          nb0 = load_b_kmaj(V, N, e_base + (t + 1) * 16, kc);
          nb1 = load_b_kmaj(V, N, e_base + (t + 1) * 16, kc + 32);
        }
        acc[t] = WMMA_BF16(pa0, vb0, acc[t]);
        acc[t] = WMMA_BF16(pa1, vb1, acc[t]);
        vb0 = nb0;
        vb1 = nb1;
      }
    }
    __syncthreads();
  }

  if (tid < 32) sLinv[tid] = 1.0f / sL[tid];
  __syncthreads();
  float li[8];
#pragma unroll
  for (int r = 0; r < 8; ++r) li[r] = sLinv[qt * 16 + 8 * hi + r];
  const int n = q0 + qt * 16 + lo;
#pragma unroll
  for (int t = 0; t < 12; ++t) {
    v8bf pk;
#pragma unroll
    for (int r = 0; r < 8; ++r) pk[r] = (__bf16)(acc[t][r] * li[r]);
    *(v8bf*)(O + (size_t)n * E + e_base + t * 16 + 8 * hi) = pk;
  }
}

// ---------------------------------------------------------------------------
// Kernel 3: output projection  Y = wo(768x768) * O^T + bo, Y stored (E,N) f32
// O stored (N,768) bf16 -> B fragments are contiguous loads (L2-resident).
// ---------------------------------------------------------------------------
template <int N>
__global__ __launch_bounds__(256) void oproj_kernel(
    const float* __restrict__ wo, const float* __restrict__ bo,
    const __bf16* __restrict__ Oall, float* __restrict__ Yall) {
  constexpr int E = 768;
  const int b = blockIdx.z;
  const __bf16* Oi = Oall + (size_t)b * N * E;
  float* Y = Yall + (size_t)b * E * N;
  const int n0 = blockIdx.x * 64;
  const int m0 = blockIdx.y * 128 + (threadIdx.x >> 5) * 16;
  const int lane = threadIdx.x & 31, lo = lane & 15, hi = lane >> 4;

  v8f acc[4] = {};
  for (int kk = 0; kk < E; kk += 32) {
    const v16bf a = load_a_f32(wo, E, m0, kk);
#pragma unroll
    for (int t = 0; t < 4; ++t) {
      const v16bf bf = load_b_kmaj(Oi, E, n0 + t * 16, kk);
      acc[t] = WMMA_BF16(a, bf, acc[t]);
    }
  }
  float bv[8];
#pragma unroll
  for (int r = 0; r < 8; ++r) bv[r] = bo[m0 + 8 * hi + r];
#pragma unroll
  for (int t = 0; t < 4; ++t) {
#pragma unroll
    for (int r = 0; r < 8; ++r)
      Y[(size_t)(m0 + 8 * hi + r) * N + n0 + t * 16 + lo] = acc[t][r] + bv[r];
  }
}

// ---------------------------------------------------------------------------
extern "C" void kernel_launch(void* const* d_in, const int* in_sizes, int n_in,
                              void* d_out, int out_size, void* d_ws, size_t ws_size,
                              hipStream_t stream) {
  (void)in_sizes; (void)n_in; (void)out_size; (void)ws_size;
  constexpr int B = 8, C = 128, N = 48 * 48, E = 768;

  const float* x  = (const float*)d_in[0];
  const float* wq = (const float*)d_in[1];
  const float* bq = (const float*)d_in[2];
  const float* wk = (const float*)d_in[3];
  const float* bk = (const float*)d_in[4];
  const float* wv = (const float*)d_in[5];
  const float* bv = (const float*)d_in[6];
  const float* wo = (const float*)d_in[7];
  const float* bo = (const float*)d_in[8];
  float* y = (float*)d_out;

  // Workspace layout (bf16): Q (B,N,C) | Kt (B,N,C) | V (B,E,N) | O (B,N,E)
  __bf16* Qb  = (__bf16*)d_ws;
  __bf16* Ktb = Qb  + (size_t)B * N * C;
  __bf16* Vb  = Ktb + (size_t)B * N * C;
  __bf16* Ob  = Vb  + (size_t)B * E * N;

  dim3 blk(256);
  proj_kernel<true,  C, N><<<dim3(N / 64, C / 128, B), blk, 0, stream>>>(wq, bq, x, Qb);
  proj_kernel<true,  C, N><<<dim3(N / 64, C / 128, B), blk, 0, stream>>>(wk, bk, x, Ktb);
  proj_kernel<false, E, N><<<dim3(N / 64, E / 128, B), blk, 0, stream>>>(wv, bv, x, Vb);
  attn_kernel<N><<<dim3(N / 32, B), blk, 0, stream>>>(Qb, Ktb, Vb, Ob);
  oproj_kernel<N><<<dim3(N / 64, E / 128, B), blk, 0, stream>>>(wo, bo, Ob, y);
}
